// Attention_57363583205911
// MI455X (gfx1250) — matrix operand order repaired and checked by reading, not compiled, not run on hardware
//
#include <hip/hip_runtime.h>

// ---------------------------------------------------------------------------
// Memory-XL attention for MI455X (gfx1250, wave32, WMMA).
// All matmuls: v_wmma_f32_16x16x32_bf16 (bf16 in, f32 accumulate).
// ---------------------------------------------------------------------------

typedef __attribute__((ext_vector_type(16))) __bf16 v16bf;
typedef __attribute__((ext_vector_type(8)))  __bf16 v8bf;
typedef __attribute__((ext_vector_type(4)))  __bf16 v4bf;
typedef __attribute__((ext_vector_type(8)))  float  v8f;
typedef __attribute__((ext_vector_type(4)))  float  v4f;

#define NB     2
#define NSEQ   2048
#define MEMLEN 2048
#define JTOT   4096
#define DIM    1024
#define INNER  1024
#define HEADS  16
#define DHEAD  64
#define QSCALE 0.125f
#define MASKV  (-3.0e38f)

static __device__ inline v16bf cat16(v8bf lo, v8bf hi) {
  v16bf r;
#pragma unroll
  for (int e = 0; e < 8; ++e) { r[e] = lo[e]; r[e + 8] = hi[e]; }
  return r;
}

static __device__ inline v8f wmma_bf(v16bf a, v16bf b, v8f c) {
  // (neg_a, A, neg_b, B, c_mod, C, reuse_a, reuse_b)
  return __builtin_amdgcn_wmma_f32_16x16x32_bf16(false, a, false, b, (short)0, c,
                                                 false, false);
}

// ---------------------------------------------------------------------------
// prep_k: mem_k f32 [b][2048][1024] -> K_all bf16 [b][4096][1024], rows 0..2047
// ---------------------------------------------------------------------------
__global__ __launch_bounds__(256) void prep_k(const float* __restrict__ mk,
                                              __bf16* __restrict__ Kall) {
  size_t idx = ((size_t)blockIdx.x * 256 + threadIdx.x) * 4;  // elem index
  v4f v = *(const v4f*)(mk + idx);
  size_t bat = idx >> 21;                     // / (2048*1024)
  size_t rem = idx & (((size_t)1 << 21) - 1);
  v4bf o;
#pragma unroll
  for (int e = 0; e < 4; ++e) o[e] = (__bf16)v[e];
  *(v4bf*)(Kall + (bat << 22) + rem) = o;     // * (4096*1024)
}

// ---------------------------------------------------------------------------
// prep_v: mem_v f32 [b][2048][1024] -> VT bf16 [b][1024][4096], cols 0..2047
// 64x64 tile transpose through LDS.
// ---------------------------------------------------------------------------
__global__ __launch_bounds__(256) void prep_v(const float* __restrict__ mv,
                                              __bf16* __restrict__ VT) {
  __shared__ __bf16 T[64][65];
  int bid = blockIdx.x;
  int jt = bid & 31, ct = (bid >> 5) & 15, bat = bid >> 9;
  int t = threadIdx.x;
  const float* src = mv + ((size_t)bat << 21) + (size_t)(jt * 64) * 1024 + ct * 64;
#pragma unroll
  for (int pass = 0; pass < 4; ++pass) {
    int row = (t >> 4) + pass * 16, cq = (t & 15) * 4;
    v4f v = *(const v4f*)(src + (size_t)row * 1024 + cq);
#pragma unroll
    for (int e = 0; e < 4; ++e) T[row][cq + e] = (__bf16)v[e];
  }
  __syncthreads();
#pragma unroll
  for (int it = 0; it < 2; ++it) {
    int ch = t + it * 256;
    int c = ch >> 3, jq = (ch & 7) * 8;
    v8bf o;
#pragma unroll
    for (int e = 0; e < 8; ++e) o[e] = T[jq + e][c];
    __bf16* dst = VT + (((size_t)bat * 1024 + ct * 64 + c) << 12) + jt * 64 + jq;
    *(v8bf*)dst = o;
  }
}

// ---------------------------------------------------------------------------
// proj_gemm: C[4096][3072] = X[4096][1024] @ [Wq | Wkv], 128x128x32 tiles.
//   cols [0,1024)    -> Qb  bf16 [b][2048][1024]  (scaled by QSCALE)
//   cols [1024,2048) -> K_all rows 2048..4095
//   cols [2048,3072) -> VT   cols 2048..4095 (transposed write via LDS)
// ---------------------------------------------------------------------------
__global__ __launch_bounds__(256) void proj_gemm(const float* __restrict__ X,
                                                 const float* __restrict__ Wq,
                                                 const float* __restrict__ Wkv,
                                                 __bf16* __restrict__ Qb,
                                                 __bf16* __restrict__ Kall,
                                                 __bf16* __restrict__ VT) {
  __shared__ __bf16 As[128][40];   // [m][k]   (row contiguous in k)
  __shared__ __bf16 Bs[128][40];   // [n][k]   (W tile, transposed)
  __shared__ float  Os[128][132];  // output staging

  int n0 = blockIdx.x * 128;
  int m0 = blockIdx.y * 128;
  int t = threadIdx.x, lane = t & 31, wave = t >> 5;
  int wm = wave >> 2, wn = wave & 3;          // 2 x 4 wave grid
  int hl = lane >> 4, l16 = lane & 15;

  const float* W; int ldw, wc0;
  if (n0 < 1024) { W = Wq;  ldw = 1024; wc0 = n0; }
  else           { W = Wkv; ldw = 2048; wc0 = n0 - 1024; }

  v8f acc[4][2] = {};

  for (int k0 = 0; k0 < DIM; k0 += 32) {
    { // A tile: X[m0..+127][k0..+31] -> bf16
      int m = t >> 1, kq = (t & 1) * 16;
      const float* src = X + (size_t)(m0 + m) * DIM + k0 + kq;
#pragma unroll
      for (int q = 0; q < 4; ++q) {
        v4f v = *(const v4f*)(src + q * 4);
#pragma unroll
        for (int e = 0; e < 4; ++e) As[m][kq + q * 4 + e] = (__bf16)v[e];
      }
    }
    { // B tile: W[k0..+31][wc0..+127] -> Bs[n][k]
      int n = t & 127, g0 = t >> 7;
      for (int g = g0; g < 8; g += 2) {
#pragma unroll
        for (int jx = 0; jx < 4; ++jx) {
          float v = W[(size_t)(k0 + g * 4 + jx) * ldw + wc0 + n];
          Bs[n][g * 4 + jx] = (__bf16)v;
        }
      }
    }
    __syncthreads();

    v16bf a[4], bfr[2];
#pragma unroll
    for (int mf = 0; mf < 4; ++mf) {
      const __bf16* p = &As[wm * 64 + mf * 16 + l16][8 * hl];
      a[mf] = cat16(*(const v8bf*)p, *(const v8bf*)(p + 16));
    }
#pragma unroll
    for (int nf = 0; nf < 2; ++nf) {
      const __bf16* p = &Bs[wn * 32 + nf * 16 + l16][8 * hl];
      bfr[nf] = cat16(*(const v8bf*)p, *(const v8bf*)(p + 16));
    }
#pragma unroll
    for (int mf = 0; mf < 4; ++mf)
#pragma unroll
      for (int nf = 0; nf < 2; ++nf)
        acc[mf][nf] = wmma_bf(a[mf], bfr[nf], acc[mf][nf]);
    __syncthreads();
  }

  // stage f32 tile in LDS
#pragma unroll
  for (int mf = 0; mf < 4; ++mf)
#pragma unroll
    for (int nf = 0; nf < 2; ++nf)
#pragma unroll
      for (int r = 0; r < 8; ++r)
        Os[wm * 64 + mf * 16 + r + 8 * hl][wn * 32 + nf * 16 + l16] = acc[mf][nf][r];
  __syncthreads();

  int bat = m0 >> 11;           // row block all in one batch (2048 % 128 == 0)
  int ir0 = m0 & 2047;
  if (n0 < 2048) {
    float sc = (n0 < 1024) ? QSCALE : 1.0f;
#pragma unroll
    for (int it = 0; it < 8; ++it) {
      int ch = t + it * 256;
      int row = ch >> 4, cq = (ch & 15) * 8;
      v8bf o;
#pragma unroll
      for (int e = 0; e < 8; ++e) o[e] = (__bf16)(Os[row][cq + e] * sc);
      __bf16* dst;
      if (n0 < 1024)
        dst = Qb + ((size_t)(bat * NSEQ + ir0 + row) * INNER) + n0 + cq;
      else
        dst = Kall + ((size_t)(bat * JTOT + MEMLEN + ir0 + row) * INNER) + (n0 - 1024) + cq;
      *(v8bf*)dst = o;
    }
  } else {
    // V: transposed write VT[bat][c][2048 + i]
#pragma unroll
    for (int it = 0; it < 8; ++it) {
      int ch = t + it * 256;
      int cc = ch >> 4, iq = (ch & 15) * 8;
      v8bf o;
#pragma unroll
      for (int e = 0; e < 8; ++e) o[e] = (__bf16)Os[iq + e][cc];
      __bf16* dst = VT + (((size_t)bat * INNER + (n0 - 2048) + cc) << 12)
                    + MEMLEN + ir0 + iq;
      *(v8bf*)dst = o;
    }
  }
}

// ---------------------------------------------------------------------------
// Flash attention. Grid = iblock(16) * h(16) * b(2), batch in the LOW grid
// bit so the two batches (which read identical pos_bias rows) run as adjacent
// co-resident blocks and share the bias stream through L2.
// Computes S^T = K Q^T so softmax rows live in lane pairs {L, L+16}, and the
// exp'd S^T C-fragments are, per lane, exactly the A-operand layout of P.
// ---------------------------------------------------------------------------
__global__ __launch_bounds__(256) void attn_fa(const __bf16* __restrict__ Qb,
                                               const __bf16* __restrict__ Kall,
                                               const __bf16* __restrict__ VT,
                                               const float* __restrict__ bias,
                                               float* __restrict__ out) {
  int bid = blockIdx.x;
  int bat = bid & 1, h = (bid >> 1) & 15, ib = bid >> 5;
  int t = threadIdx.x, lane = t & 31, wave = t >> 5;
  int hl = lane >> 4, l16 = lane & 15;
  int i0 = ib * 128 + wave * 16;
  int icol = i0 + l16;                       // softmax row tracked by this lane

  // B = Q^T fragments (32d x 16i), two d-halves
  v16bf b_q[2];
  const __bf16* qrow = Qb + ((size_t)(bat * NSEQ + icol) * INNER) + h * DHEAD;
#pragma unroll
  for (int dh = 0; dh < 2; ++dh) {
    const __bf16* p = qrow + dh * 32 + 8 * hl;
    b_q[dh] = cat16(*(const v8bf*)p, *(const v8bf*)(p + 16));
  }

  v8f acc[4] = {};
  float m = -1e30f, l = 0.0f;
  const float* brow = bias + ((size_t)h * NSEQ + icol) * JTOT;

  for (int j0 = 0; j0 < i0 + 2064; j0 += 32) {
    // prefetch next iteration's bias cachelines (global_prefetch_b8)
    __builtin_prefetch(brow + j0 + 32 + 8 * hl, 0, 1);
    // ---- S^T = K * Q^T -------------------------------------------------
    v8f st[2];
#pragma unroll
    for (int jf = 0; jf < 2; ++jf) {
      const __bf16* krow =
          Kall + ((size_t)(bat * JTOT + j0 + jf * 16 + l16) * INNER) + h * DHEAD;
      v8f c = {};
#pragma unroll
      for (int dh = 0; dh < 2; ++dh) {
        const __bf16* p = krow + dh * 32 + 8 * hl;
        v16bf a_k = cat16(*(const v8bf*)p, *(const v8bf*)(p + 16));
        c = wmma_bf(a_k, b_q[dh], c);
      }
      st[jf] = c;
    }
    // ---- bias + causal mask (allowed iff jj <= icol + 2048) ------------
    float s[2][8];
#pragma unroll
    for (int jf = 0; jf < 2; ++jf) {
      int jj0 = j0 + jf * 16 + 8 * hl;
      v4f b0 = *(const v4f*)(brow + jj0);
      v4f b1 = *(const v4f*)(brow + jj0 + 4);
#pragma unroll
      for (int r = 0; r < 8; ++r) {
        float bv = (r < 4) ? b0[r] : b1[r - 4];
        float v = st[jf][r] + bv;
        s[jf][r] = ((jj0 + r) > (icol + MEMLEN)) ? MASKV : v;
      }
    }
    // ---- online softmax (row = lane pair {l16, l16+16}) ----------------
    float tmax = s[0][0];
#pragma unroll
    for (int jf = 0; jf < 2; ++jf)
#pragma unroll
      for (int r = 0; r < 8; ++r) tmax = fmaxf(tmax, s[jf][r]);
    tmax = fmaxf(tmax, __shfl_xor(tmax, 16, 32));
    float newm = fmaxf(m, tmax);
    float corr = __expf(m - newm);
    float rsum = 0.0f;
#pragma unroll
    for (int jf = 0; jf < 2; ++jf)
#pragma unroll
      for (int r = 0; r < 8; ++r) {
        float p = __expf(s[jf][r] - newm);
        s[jf][r] = p;
        rsum += p;
      }
    rsum += __shfl_xor(rsum, 16, 32);
    l = l * corr + rsum;
    m = newm;
    // ---- P^T C-frags == A-layout of P (16i x 32j), per-lane cast -------
    v16bf a_p;
#pragma unroll
    for (int e = 0; e < 8; ++e) {
      a_p[e] = (__bf16)s[0][e];
      a_p[e + 8] = (__bf16)s[1][e];
    }
    // rescale accumulators: out-frag row i = r + 8*hl, corr held by lane i
    float cr[8];
#pragma unroll
    for (int r = 0; r < 8; ++r) cr[r] = __shfl(corr, r + 8 * hl, 32);
#pragma unroll
    for (int df = 0; df < 4; ++df)
#pragma unroll
      for (int r = 0; r < 8; ++r) acc[df][r] *= cr[r];
    // ---- acc += P @ V (B from VT, contiguous along j) ------------------
#pragma unroll
    for (int df = 0; df < 4; ++df) {
      const __bf16* vp =
          VT + (((size_t)bat * INNER + h * DHEAD + df * 16 + l16) << 12)
          + j0 + 8 * hl;
      v16bf b_v = cat16(*(const v8bf*)vp, *(const v8bf*)(vp + 16));
      acc[df] = wmma_bf(a_p, b_v, acc[df]);
    }
  }

  // ---- epilogue: out[bat][i][h*64 + d] = acc * (1/l) -------------------
  float li[8];
#pragma unroll
  for (int r = 0; r < 8; ++r) li[r] = __frcp_rn(__shfl(l, r + 8 * hl, 32));
#pragma unroll
  for (int df = 0; df < 4; ++df)
#pragma unroll
    for (int r = 0; r < 8; ++r) {
      int irow = i0 + r + 8 * hl;
      out[((size_t)(bat * NSEQ + irow) * INNER) + h * DHEAD + df * 16 + l16] =
          acc[df][r] * li[r];
    }
}

// ---------------------------------------------------------------------------
extern "C" void kernel_launch(void* const* d_in, const int* in_sizes, int n_in,
                              void* d_out, int out_size, void* d_ws, size_t ws_size,
                              hipStream_t stream) {
  const float* x        = (const float*)d_in[0];
  const float* mem_k    = (const float*)d_in[1];
  const float* mem_v    = (const float*)d_in[2];
  const float* pos_bias = (const float*)d_in[3];
  const float* Wq       = (const float*)d_in[4];
  const float* Wkv      = (const float*)d_in[5];
  float* out = (float*)d_out;

  // workspace: Qb (8MB) | K_all (16MB) | VT (16MB), bf16
  __bf16* Qb   = (__bf16*)d_ws;
  __bf16* Kall = Qb + (size_t)NB * NSEQ * INNER;
  __bf16* VT   = Kall + (size_t)NB * JTOT * INNER;

  prep_k<<<4096, 256, 0, stream>>>(mem_k, Kall);
  prep_v<<<1024, 256, 0, stream>>>(mem_v, VT);
  dim3 gg(24, 32);  // 3072/128 n-tiles  x  4096/128 m-tiles
  proj_gemm<<<gg, 256, 0, stream>>>(x, Wq, Wkv, Qb, Kall, VT);
  attn_fa<<<512, 256, 0, stream>>>(Qb, Kall, VT, pos_bias, out);
}
